// HippocampusModule_30837865185416
// MI455X (gfx1250) — matrix operand order repaired and checked by reading, not compiled, not run on hardware
//
#include <hip/hip_runtime.h>
#include <hip/hip_bf16.h>
#include <math.h>

// ---------------------------------------------------------------------------
// HippocampusModule forward for MI455X (gfx1250, wave32, WMMA bf16).
//
// Roofline: ~1.2 TFLOP total, ~92% in the two memory-bank GEMMs (K/V = 64K x
// 1024). 550 MB of fp32 parameters -> ~24us of HBM time at 23.3 TB/s, so the
// pipeline is compute-bound: everything runs through
// v_wmma_f32_16x16x32_bf16 with f32 accumulation. The softmax over 65536
// logits per query is fused flash-style (online max/sum) so the 1 GB logits
// matrix never touches memory. Dense GEMM tiles are staged with the Tensor
// Data Mover (tensor_load_to_lds + s_wait_tensorcnt), double-buffered so the
// DMA overlaps the WMMA stream.
// ---------------------------------------------------------------------------

typedef __bf16 bf16_t;
typedef __attribute__((ext_vector_type(16))) __bf16 v16bf;
typedef __attribute__((ext_vector_type(8)))  float  v8f;
typedef __attribute__((ext_vector_type(4)))  unsigned int u32x4_t;
typedef __attribute__((ext_vector_type(8)))  int i32x8_t;
typedef __attribute__((ext_vector_type(4)))  int i32x4_t;

#define EMB 768
#define HID 1024
#define MEMN 65536
#define NQ  4096

#if defined(__has_builtin)
#if __has_builtin(__builtin_amdgcn_tensor_load_to_lds) && \
    __has_builtin(__builtin_amdgcn_s_wait_tensorcnt)
#define USE_TDM 1
#endif
#endif
#ifndef USE_TDM
#define USE_TDM 0
#endif

__device__ __forceinline__ bf16_t f2bf(float f) { return (bf16_t)f; }

__device__ __forceinline__ float gelu_erf(float x) {
  return 0.5f * x * (1.0f + erff(x * 0.70710678118654752f));
}

__device__ __forceinline__ v8f wmma_bf16(v16bf a, v16bf b, v8f c) {
  // 8-arg probe-confirmed signature: (neg_a, A, neg_b, B, c_mod, C, reuse_a, reuse_b)
  return __builtin_amdgcn_wmma_f32_16x16x32_bf16(false, a, false, b, (short)0, c,
                                                 false, false);
}

// A-matrix fragment (16x32 bf16, M x K), source row-major with row stride `ld`
// (elements). Layout: lanes 0-15 -> M=lane, K 0..15;
// lanes 16-31 -> M=lane-16, K 16..31. Two 16B loads per lane.
__device__ __forceinline__ v16bf ldfrag_A(const bf16_t* base, int ld, int m0,
                                          int k0, int lane) {
  int r  = m0 + (lane & 15);
  int kb = k0 + ((lane & 16) ? 16 : 0);
  const uint4* p = reinterpret_cast<const uint4*>(base + (long long)r * ld + kb);
  union { v16bf v; uint4 u[2]; } x;
  x.u[0] = p[0];   // K kb .. kb+7
  x.u[1] = p[1];   // K kb+8 .. kb+15
  return x.v;
}

// B-matrix fragment (32x16 bf16, K x N), source is B-transposed [N][K] with row
// stride `ld`. Layout (contiguous k, same as ldfrag_A): lanes 0-15 -> N=lane, K 0..15; lanes 16-31 ->
// N=lane-16, K 16..31. 32 contiguous bytes per lane.
__device__ __forceinline__ v16bf ldfrag_B(const bf16_t* base, long long ld,
                                          int n0, int k0, int lane) {
  int r  = n0 + (lane & 15);
  int kb = k0 + ((lane & 16) ? 16 : 0);
  const uint4* p = reinterpret_cast<const uint4*>(base + (long long)r * ld + kb);
  union { v16bf v; uint4 u[2]; } x;
  x.u[0] = p[0];
  x.u[1] = p[1];
  return x.v;
}

__device__ __forceinline__ float rmax16(float v) {
#pragma unroll
  for (int off = 8; off > 0; off >>= 1) v = fmaxf(v, __shfl_xor(v, off, 16));
  return v;
}
__device__ __forceinline__ float rsum16(float v) {
#pragma unroll
  for (int off = 8; off > 0; off >>= 1) v += __shfl_xor(v, off, 16);
  return v;
}

#if USE_TDM
// Issue a TDM 2-D tile load (bf16 elements) into LDS. D# packing per CDNA5 ISA
// sec. 8.3/8.4: group0 = {count/type, lds_addr, global_addr}, group1 =
// {data_size, tensor dims, tile dims, dim0 stride}. Uniform (SGPR) operands;
// one issue per wave; tracked on TENSORcnt. This toolchain exposes the 6-arg
// builtin (g0, g1, g2, g3, g4, cpol); groups 2-4 are unused for 2-D tiles.
__device__ __forceinline__ void tdm_load_2d(unsigned lds_off_bytes,
                                            const bf16_t* gsrc,
                                            unsigned tile_x,     // elems (contig)
                                            unsigned tile_y,     // rows
                                            unsigned row_stride) // elems
{
  unsigned long long ga = (unsigned long long)(uintptr_t)gsrc;
  u32x4_t g0;
  g0[0] = 1u;                                        // count=1 (valid), user mode
  g0[1] = lds_off_bytes;                             // lds_addr
  g0[2] = (unsigned)(ga & 0xFFFFFFFFu);              // global_addr[31:0]
  g0[3] = (unsigned)((ga >> 32) & 0x01FFFFFFu)       // global_addr[56:32]
        | (2u << 30);                                // type = 2 ("image")
  const unsigned td0 = row_stride;                   // tensor_dim0 (>= tile_x)
  const unsigned td1 = 0x40000000u;                  // tensor_dim1: huge
  const unsigned long long s0 = row_stride;          // tensor_dim0_stride
  i32x8_t g1;
  g1[0] = (int)(1u << 16);                           // data_size = 1 -> 2 bytes
  g1[1] = (int)((td0 & 0xFFFFu) << 16);              // dim0[15:0]
  g1[2] = (int)(((td0 >> 16) & 0xFFFFu) | ((td1 & 0xFFFFu) << 16));
  g1[3] = (int)(((td1 >> 16) & 0xFFFFu) | (tile_x << 16));  // tile_dim0
  g1[4] = (int)(tile_y & 0xFFFFu);                   // tile_dim1 (tile_dim2=0)
  g1[5] = (int)(unsigned)(s0 & 0xFFFFFFFFu);         // dim0 stride [31:0]
  g1[6] = (int)(unsigned)((s0 >> 32) & 0xFFFFu);     // stride[47:32], d1stride=0
  g1[7] = 0;
  i32x4_t z4 = {0, 0, 0, 0};
  i32x8_t z8 = {0, 0, 0, 0, 0, 0, 0, 0};
  __builtin_amdgcn_tensor_load_to_lds(g0, g1, z4, z4, z8, 0);
}
#endif

// ---------------------------------------------------------------------------
// f32 -> bf16 flat convert
// ---------------------------------------------------------------------------
__global__ void k_cvt(const float* __restrict__ x, bf16_t* __restrict__ y,
                      long long n) {
  long long i = (long long)blockIdx.x * blockDim.x + threadIdx.x;
  long long stride = (long long)gridDim.x * blockDim.x;
  for (; i < n; i += stride) y[i] = f2bf(x[i]);
}

// ---------------------------------------------------------------------------
// f32 [K][N] -> bf16 [N][K] tiled transpose (32x32 tiles via LDS)
// ---------------------------------------------------------------------------
__global__ void k_transpose_cvt(const float* __restrict__ W,
                                bf16_t* __restrict__ Wt, int K, int N) {
  __shared__ float t[32][33];
  int kb = blockIdx.y * 32, nb = blockIdx.x * 32;
  int tx = threadIdx.x, ty = threadIdx.y;
#pragma unroll
  for (int r = ty; r < 32; r += 8)
    t[r][tx] = W[(long long)(kb + r) * N + (nb + tx)];
  __syncthreads();
#pragma unroll
  for (int r = ty; r < 32; r += 8)
    Wt[(long long)(nb + r) * K + (kb + tx)] = f2bf(t[tx][r]);
}

// ---------------------------------------------------------------------------
// WMMA GEMM: C[M][N] = act(A[M][K](bf16) * Bt[N][K](bf16)^T + bias)
// Block = 256 threads = 8 waves; 128x128 output tile; wave grid 4(M) x 2(N),
// each wave owns a 32x64 sub-tile = 2x4 C fragments (64 acc VGPRs).
// Tile staging: TDM double-buffered (DMA overlaps WMMA) when available.
// ---------------------------------------------------------------------------
template <int ACT>
__global__ void __launch_bounds__(256, 2)
k_gemm(const bf16_t* __restrict__ A, int lda,
       const bf16_t* __restrict__ Bt, int ldb,
       const float* __restrict__ bias,
       bf16_t* __restrict__ Cbf, int ldcb,
       float* __restrict__ Cf, int ldcf,
       int K) {
#if USE_TDM
  __shared__ bf16_t sA[2][128 * 32];
  __shared__ bf16_t sB[2][128 * 32];
#else
  __shared__ bf16_t sA[1][128 * 32];
  __shared__ bf16_t sB[1][128 * 32];
#endif
  const int tid = threadIdx.x, lane = tid & 31, wid = tid >> 5;
  const int wm = wid & 3, wn = wid >> 2;
  const int row0 = blockIdx.y * 128, col0 = blockIdx.x * 128;

  v8f acc[2][4];
#pragma unroll
  for (int mt = 0; mt < 2; ++mt)
#pragma unroll
    for (int nt = 0; nt < 4; ++nt)
      acc[mt][nt] = (v8f){0.f, 0.f, 0.f, 0.f, 0.f, 0.f, 0.f, 0.f};

#if USE_TDM
  // prologue: DMA first k-slice into buffer 0
  if (wid == 0) {
    tdm_load_2d((unsigned)(uintptr_t)&sA[0][0], A + (long long)row0 * lda, 32,
                128, (unsigned)lda);
    tdm_load_2d((unsigned)(uintptr_t)&sB[0][0], Bt + (long long)col0 * ldb, 32,
                128, (unsigned)ldb);
  }
  int buf = 0;
  for (int k0 = 0; k0 < K; k0 += 32) {
    if (wid == 0) {
      if (k0 + 32 < K) {  // kick DMA for the next slice, then wait for current
        tdm_load_2d((unsigned)(uintptr_t)&sA[buf ^ 1][0],
                    A + (long long)row0 * lda + (k0 + 32), 32, 128,
                    (unsigned)lda);
        tdm_load_2d((unsigned)(uintptr_t)&sB[buf ^ 1][0],
                    Bt + (long long)col0 * ldb + (k0 + 32), 32, 128,
                    (unsigned)ldb);
        __builtin_amdgcn_s_wait_tensorcnt(2);
      } else {
        __builtin_amdgcn_s_wait_tensorcnt(0);
      }
    }
    __syncthreads();
    v16bf a0 = ldfrag_A(sA[buf], 32, wm * 32, 0, lane);
    v16bf a1 = ldfrag_A(sA[buf], 32, wm * 32 + 16, 0, lane);
#pragma unroll
    for (int nt = 0; nt < 4; ++nt) {
      v16bf b = ldfrag_B(sB[buf], 32, wn * 64 + nt * 16, 0, lane);
      acc[0][nt] = wmma_bf16(a0, b, acc[0][nt]);
      acc[1][nt] = wmma_bf16(a1, b, acc[1][nt]);
    }
    __syncthreads();
    buf ^= 1;
  }
#else
  uint4* sa4 = reinterpret_cast<uint4*>(&sA[0][0]);
  uint4* sb4 = reinterpret_cast<uint4*>(&sB[0][0]);
  for (int k0 = 0; k0 < K; k0 += 32) {
#pragma unroll
    for (int i = tid; i < 512; i += 256) {
      int r = i >> 2, c = i & 3;
      sa4[i] = *reinterpret_cast<const uint4*>(A + (long long)(row0 + r) * lda + k0 + c * 8);
      sb4[i] = *reinterpret_cast<const uint4*>(Bt + (long long)(col0 + r) * ldb + k0 + c * 8);
    }
    __syncthreads();
    v16bf a0 = ldfrag_A(sA[0], 32, wm * 32, 0, lane);
    v16bf a1 = ldfrag_A(sA[0], 32, wm * 32 + 16, 0, lane);
#pragma unroll
    for (int nt = 0; nt < 4; ++nt) {
      v16bf b = ldfrag_B(sB[0], 32, wn * 64 + nt * 16, 0, lane);
      acc[0][nt] = wmma_bf16(a0, b, acc[0][nt]);
      acc[1][nt] = wmma_bf16(a1, b, acc[1][nt]);
    }
    __syncthreads();
  }
#endif

  // epilogue: C layout -> lane = N col, VGPR i = M row (i / i+8 per lane half)
  const int half8 = (lane & 16) ? 8 : 0;
#pragma unroll
  for (int nt = 0; nt < 4; ++nt) {
    int col = col0 + wn * 64 + nt * 16 + (lane & 15);
    float bv = bias ? bias[col] : 0.f;
#pragma unroll
    for (int mt = 0; mt < 2; ++mt)
#pragma unroll
      for (int i = 0; i < 8; ++i) {
        int row = row0 + wm * 32 + mt * 16 + i + half8;
        float v = acc[mt][nt][i] + bv;
        if (ACT == 1) v = gelu_erf(v);
        if (Cbf) Cbf[(long long)row * ldcb + col] = f2bf(v);
        if (Cf)  Cf[(long long)row * ldcf + col] = v;
      }
  }
}

// ---------------------------------------------------------------------------
// Fused attention over memory bank (flash-style, online softmax):
//   retrieved = softmax(Q * Kmem^T / 32) * V
// Block = 256 threads = 8 waves, 32 query rows per block (grid = 128).
// Per 128-row memory tile: wave w computes logits for mem cols [16w,16w+16)
// (64 QK WMMAs), then owns HID output cols [128w,128w+128) (64 PV WMMAs).
// B fragments stream from global with a depth-2 rotating prefetch so 4
// global_load_b128 stay in flight ahead of each WMMA pair.
// ---------------------------------------------------------------------------
__global__ void __launch_bounds__(256, 1)
k_attention(const bf16_t* __restrict__ Q,     // [NQ][HID] (the "key" proj)
            const bf16_t* __restrict__ Km,    // [MEMN][HID]  (B layout)
            const bf16_t* __restrict__ Vt,    // [HID][MEMN]  (B layout)
            bf16_t* __restrict__ O, int ldo)  // [NQ][ldo]
{
  __shared__ bf16_t sQ[32 * HID];   // 64 KB
  __shared__ bf16_t sP[32 * 128];   // 8 KB
  __shared__ float sm[32], sl[32], snew[32], sfac[32];
  __shared__ float swmax[8][32], swsum[8][32];

  const int tid = threadIdx.x, lane = tid & 31, wid = tid >> 5;
  const int q0 = blockIdx.x * 32;
  const int half8 = (lane & 16) ? 8 : 0;
  const float sc = 0.03125f;  // 1/sqrt(1024)

  {  // stage Q rows (contiguous 64 KB copy)
    const uint4* gq = reinterpret_cast<const uint4*>(Q + (long long)q0 * HID);
    uint4* sq4 = reinterpret_cast<uint4*>(sQ);
    for (int i = tid; i < 32 * HID / 8; i += 256) sq4[i] = gq[i];
  }
  if (tid < 32) { sm[tid] = -1e30f; sl[tid] = 0.f; }

  v8f acc[2][8];
#pragma unroll
  for (int mt = 0; mt < 2; ++mt)
#pragma unroll
    for (int nt = 0; nt < 8; ++nt)
      acc[mt][nt] = (v8f){0.f, 0.f, 0.f, 0.f, 0.f, 0.f, 0.f, 0.f};
  __syncthreads();

  for (int it = 0; it < MEMN; it += 128) {
    // hint the next Km tile toward L2 while computing this one
    if (it + 128 < MEMN)
      __builtin_prefetch(Km + (long long)(it + 128 + wid * 16 + (lane & 15)) * HID, 0, 1);

    // ---- QK: 32x16 logit tile for this wave's 16 memory columns
    const int n0 = it + wid * 16;
    v8f c0 = (v8f){0.f, 0.f, 0.f, 0.f, 0.f, 0.f, 0.f, 0.f};
    v8f c1 = c0;
    v16bf kb0 = ldfrag_B(Km, HID, n0, 0, lane);
    v16bf kb1 = ldfrag_B(Km, HID, n0, 32, lane);
#pragma unroll 8
    for (int kt = 0; kt < HID; kt += 32) {
      v16bf bc = kb0;
      kb0 = kb1;
      int kn = (kt + 64 < HID) ? kt + 64 : kt;  // branchless clamp
      kb1 = ldfrag_B(Km, HID, n0, kn, lane);
      v16bf a0 = ldfrag_A(sQ, HID, 0,  kt, lane);
      v16bf a1 = ldfrag_A(sQ, HID, 16, kt, lane);
      c0 = wmma_bf16(a0, bc, c0);
      c1 = wmma_bf16(a1, bc, c1);
    }

    // ---- per-row tile max (across this wave's 16 cols), publish per wave
#pragma unroll
    for (int i = 0; i < 8; ++i) {
      float v0 = rmax16(c0[i] * sc);
      float v1 = rmax16(c1[i] * sc);
      if ((lane & 15) == 0) {
        swmax[wid][i + half8]      = v0;
        swmax[wid][16 + i + half8] = v1;
      }
    }
    __syncthreads();
    if (tid < 32) {
      float nm = sm[tid];
#pragma unroll
      for (int w = 0; w < 8; ++w) nm = fmaxf(nm, swmax[w][tid]);
      snew[tid] = nm;
      sfac[tid] = __expf(sm[tid] - nm);
      sm[tid]   = nm;
    }
    __syncthreads();

    // ---- P = exp(logit - newmax), staged to LDS; partial row sums
#pragma unroll
    for (int i = 0; i < 8; ++i) {
      int r0 = i + half8, r1 = 16 + i + half8;
      float p0 = __expf(c0[i] * sc - snew[r0]);
      float p1 = __expf(c1[i] * sc - snew[r1]);
      sP[r0 * 128 + wid * 16 + (lane & 15)] = f2bf(p0);
      sP[r1 * 128 + wid * 16 + (lane & 15)] = f2bf(p1);
      float s0 = rsum16(p0), s1 = rsum16(p1);
      if ((lane & 15) == 0) { swsum[wid][r0] = s0; swsum[wid][r1] = s1; }
    }
    // ---- rescale running accumulators by exp(oldmax - newmax)
#pragma unroll
    for (int i = 0; i < 8; ++i) {
      float f0 = sfac[i + half8];
      float f1 = sfac[16 + i + half8];
#pragma unroll
      for (int nt = 0; nt < 8; ++nt) {
        acc[0][nt][i] *= f0;
        acc[1][nt][i] *= f1;
      }
    }
    __syncthreads();
    if (tid < 32) {
      float s = 0.f;
#pragma unroll
      for (int w = 0; w < 8; ++w) s += swsum[w][tid];
      sl[tid] = sl[tid] * sfac[tid] + s;
    }

    // ---- PV: acc += P(32x128) * V(128 x [this wave's 128 HID cols])
    v16bf pa[2][4];
#pragma unroll
    for (int mt = 0; mt < 2; ++mt)
#pragma unroll
      for (int kt = 0; kt < 4; ++kt)
        pa[mt][kt] = ldfrag_A(sP, 128, mt * 16, kt * 32, lane);

    const int vrow0 = wid * 128;
    v16bf vb0 = ldfrag_B(Vt, (long long)MEMN, vrow0, it, lane);
    v16bf vb1 = ldfrag_B(Vt, (long long)MEMN, vrow0, it + 32, lane);
#pragma unroll
    for (int s = 0; s < 32; ++s) {
      const int nt = s >> 2, kt = s & 3;
      v16bf bc = vb0;
      vb0 = vb1;
      const int s2 = (s + 2 < 32) ? s + 2 : s;  // branchless clamp
      vb1 = ldfrag_B(Vt, (long long)MEMN, vrow0 + ((s2 >> 2) << 4),
                     it + ((s2 & 3) << 5), lane);
      acc[0][nt] = wmma_bf16(pa[0][kt], bc, acc[0][nt]);
      acc[1][nt] = wmma_bf16(pa[1][kt], bc, acc[1][nt]);
    }
    __syncthreads();
  }

  // ---- normalize and emit bf16 into combined[:, 1024:2048]
#pragma unroll
  for (int mt = 0; mt < 2; ++mt)
#pragma unroll
    for (int i = 0; i < 8; ++i) {
      int r = mt * 16 + i + half8;
      float invl = 1.f / sl[r];
#pragma unroll
      for (int nt = 0; nt < 8; ++nt) {
        int col = wid * 128 + nt * 16 + (lane & 15);
        O[(long long)(q0 + r) * ldo + col] = f2bf(acc[mt][nt][i] * invl);
      }
    }
}

// ---------------------------------------------------------------------------
// Host-side orchestration
// ---------------------------------------------------------------------------
static inline size_t alignup(size_t x) { return (x + 255) & ~(size_t)255; }

extern "C" void kernel_launch(void* const* d_in, const int* in_sizes, int n_in,
                              void* d_out, int out_size, void* d_ws, size_t ws_size,
                              hipStream_t stream) {
  const float* query   = (const float*)d_in[0];
  const float* W_in    = (const float*)d_in[1];
  const float* b_in    = (const float*)d_in[2];
  const float* W_e1    = (const float*)d_in[3];
  const float* b_e1    = (const float*)d_in[4];
  const float* W_e2    = (const float*)d_in[5];
  const float* b_e2    = (const float*)d_in[6];
  const float* W_k     = (const float*)d_in[7];
  const float* b_k     = (const float*)d_in[8];
  const float* mem_k   = (const float*)d_in[9];
  const float* mem_v   = (const float*)d_in[10];
  const float* W_r1    = (const float*)d_in[11];
  const float* b_r1    = (const float*)d_in[12];
  const float* W_r2    = (const float*)d_in[13];
  const float* b_r2    = (const float*)d_in[14];
  float* out = (float*)d_out;

  // carve bf16 workspace
  char* p = (char*)d_ws;
  auto carve = [&](size_t elems) {
    bf16_t* r = (bf16_t*)p;
    p += alignup(elems * sizeof(bf16_t));
    return r;
  };
  bf16_t* q_bf   = carve((size_t)NQ * EMB);
  bf16_t* Wt_in  = carve((size_t)HID * EMB);          // [1024][768]
  bf16_t* Wt_e1  = carve((size_t)2 * HID * HID);      // [2048][1024]
  bf16_t* Wt_e2  = carve((size_t)HID * 2 * HID);      // [1024][2048]
  bf16_t* Wt_k   = carve((size_t)HID * HID);          // [1024][1024]
  bf16_t* Wt_r1  = carve((size_t)2 * HID * 2 * HID);  // [2048][2048]
  bf16_t* Wt_r2  = carve((size_t)EMB * 2 * HID);      // [768][2048]
  bf16_t* km_bf  = carve((size_t)MEMN * HID);         // [65536][1024]
  bf16_t* vt_bf  = carve((size_t)HID * MEMN);         // [1024][65536]
  bf16_t* x_bf   = carve((size_t)NQ * HID);
  bf16_t* h_bf   = carve((size_t)NQ * 2 * HID);
  bf16_t* comb   = carve((size_t)NQ * 2 * HID);       // [enc | retrieved]
  bf16_t* key_bf = carve((size_t)NQ * HID);
  bf16_t* r_bf   = carve((size_t)NQ * 2 * HID);
  (void)ws_size; (void)n_in; (void)in_sizes; (void)out_size;

  const dim3 tb(32, 8);

  // precision staging: flat converts + transposed-weight converts
  k_cvt<<<512, 256, 0, stream>>>(query, q_bf, (long long)NQ * EMB);
  k_cvt<<<4096, 256, 0, stream>>>(mem_k, km_bf, (long long)MEMN * HID);
  k_transpose_cvt<<<dim3(HID / 32, EMB / 32), tb, 0, stream>>>(W_in, Wt_in, EMB, HID);
  k_transpose_cvt<<<dim3(2 * HID / 32, HID / 32), tb, 0, stream>>>(W_e1, Wt_e1, HID, 2 * HID);
  k_transpose_cvt<<<dim3(HID / 32, 2 * HID / 32), tb, 0, stream>>>(W_e2, Wt_e2, 2 * HID, HID);
  k_transpose_cvt<<<dim3(HID / 32, HID / 32), tb, 0, stream>>>(W_k, Wt_k, HID, HID);
  k_transpose_cvt<<<dim3(2 * HID / 32, 2 * HID / 32), tb, 0, stream>>>(W_r1, Wt_r1, 2 * HID, 2 * HID);
  k_transpose_cvt<<<dim3(EMB / 32, 2 * HID / 32), tb, 0, stream>>>(W_r2, Wt_r2, 2 * HID, EMB);
  k_transpose_cvt<<<dim3(HID / 32, MEMN / 32), tb, 0, stream>>>(mem_v, vt_bf, MEMN, HID);

  // 1) x = query @ W_in + b_in
  k_gemm<0><<<dim3(HID / 128, NQ / 128), 256, 0, stream>>>(
      q_bf, EMB, Wt_in, EMB, b_in, x_bf, HID, (float*)nullptr, 0, EMB);
  // 2) h = gelu(x @ W_e1 + b_e1)
  k_gemm<1><<<dim3(2 * HID / 128, NQ / 128), 256, 0, stream>>>(
      x_bf, HID, Wt_e1, HID, b_e1, h_bf, 2 * HID, (float*)nullptr, 0, HID);
  // 3) encoded = h @ W_e2 + b_e2  -> combined[:, 0:1024]
  k_gemm<0><<<dim3(HID / 128, NQ / 128), 256, 0, stream>>>(
      h_bf, 2 * HID, Wt_e2, 2 * HID, b_e2, comb, 2 * HID, (float*)nullptr, 0, 2 * HID);
  // 4) key = encoded @ W_k + b_k
  k_gemm<0><<<dim3(HID / 128, NQ / 128), 256, 0, stream>>>(
      comb, 2 * HID, Wt_k, HID, b_k, key_bf, HID, (float*)nullptr, 0, HID);
  // 5-7) fused softmax(key @ mem_keys^T / 32) @ mem_values -> combined[:, 1024:]
  k_attention<<<NQ / 32, 256, 0, stream>>>(key_bf, km_bf, vt_bf,
                                           comb + HID, 2 * HID);
  // 8-9) r = gelu(combined @ W_r1 + b_r1)
  k_gemm<1><<<dim3(2 * HID / 128, NQ / 128), 256, 0, stream>>>(
      comb, 2 * HID, Wt_r1, 2 * HID, b_r1, r_bf, 2 * HID, (float*)nullptr, 0, 2 * HID);
  // 10) result = r @ W_r2 + b_r2 (f32 out)
  k_gemm<0><<<dim3(EMB / 128, NQ / 128), 256, 0, stream>>>(
      r_bf, 2 * HID, Wt_r2, 2 * HID, b_r2, (bf16_t*)nullptr, 0, out, EMB, 2 * HID);
}
